// RippleHeadExtrapolatable_18202071400559
// MI455X (gfx1250) — matrix operand order repaired and checked by reading, not compiled, not run on hardware
//
#include <hip/hip_runtime.h>
#include <hip/hip_bf16.h>
#include <math.h>

typedef __attribute__((ext_vector_type(16))) __bf16 v16bf;
typedef __attribute__((ext_vector_type(8)))  __bf16 v8bf;
typedef __attribute__((ext_vector_type(8)))  float  v8f;

#define T_SEQ 4096
#define C_EMB 1024
#define HD    64
#define BATCH 4

// ---------------------------------------------------------------------------
// Kernel 0: W[c][n] (fp32, [1024x64]) -> wt[mi][n][c] (bf16, [3][64][1024])
// Row-per-output-column layout makes WMMA B-fragments contiguous 32B loads.
// ---------------------------------------------------------------------------
__global__ void wprep_kernel(const float* __restrict__ Wq,
                             const float* __restrict__ Wk,
                             const float* __restrict__ Wv,
                             __bf16* __restrict__ wt) {
  int idx = blockIdx.x * blockDim.x + threadIdx.x;
  if (idx >= 3 * HD * C_EMB) return;
  int mi  = idx / (HD * C_EMB);
  int rem = idx - mi * (HD * C_EMB);
  int n   = rem / C_EMB;
  int c   = rem - n * C_EMB;
  const float* W = (mi == 0) ? Wq : (mi == 1) ? Wk : Wv;
  wt[idx] = (__bf16)W[c * HD + n];
}

// ---------------------------------------------------------------------------
// Kernel 1: FUSED projections. One pass over x computes q, k, v:
//   each A-fragment of x feeds 3 WMMAs (Wq, Wk, Wv) -> x read once per wave
//   instead of once per (wave, matrix).
// q = (x@Wq)*H^-0.5, k = x@Wk  (bf16 row-major [B*T,64]);
// v stored transposed vt[b][h][t] (bf16) so phase-2 B-frags are contiguous.
// grid = (B*T/16), block = 128 (4 waves), wave w -> output cols [16w,16w+16).
// Inner loop: 3 x v_wmma_f32_16x16x32_bf16 per 32-wide K step (96 per wave).
// ---------------------------------------------------------------------------
__global__ __launch_bounds__(128)
void proj_kernel(const float* __restrict__ x, const __bf16* __restrict__ wt,
                 __bf16* __restrict__ qbf, __bf16* __restrict__ kbf,
                 __bf16* __restrict__ vtbf) {
  const int mtile = blockIdx.x;
  const int wave  = threadIdx.x >> 5;
  const int lane  = threadIdx.x & 31;
  const int ln    = lane & 15;             // N index / A-row index
  const int lh    = lane >> 4;             // half-wave select
  const int n0    = wave * 16;
  const int m0    = mtile * 16;

  const float*  xr = x + (size_t)(m0 + ln) * C_EMB;
  const __bf16* wq = wt + 0 * (size_t)HD * C_EMB + (size_t)(n0 + ln) * C_EMB;
  const __bf16* wk = wt + 1 * (size_t)HD * C_EMB + (size_t)(n0 + ln) * C_EMB;
  const __bf16* wv = wt + 2 * (size_t)HD * C_EMB + (size_t)(n0 + ln) * C_EMB;

  v8f accq = {}, acck = {}, accv = {};
  for (int k0 = 0; k0 < C_EMB; k0 += 32) {
    // A fragment (x row m0+ln): K = k0 + lh*16 + e
    const float* ap = xr + k0 + lh * 16;
    float4 f0 = *(const float4*)(ap);
    float4 f1 = *(const float4*)(ap + 4);
    float4 f2 = *(const float4*)(ap + 8);
    float4 f3 = *(const float4*)(ap + 12);
    v16bf a;
    a[0]  = (__bf16)f0.x; a[1]  = (__bf16)f0.y; a[2]  = (__bf16)f0.z; a[3]  = (__bf16)f0.w;
    a[4]  = (__bf16)f1.x; a[5]  = (__bf16)f1.y; a[6]  = (__bf16)f1.z; a[7]  = (__bf16)f1.w;
    a[8]  = (__bf16)f2.x; a[9]  = (__bf16)f2.y; a[10] = (__bf16)f2.z; a[11] = (__bf16)f2.w;
    a[12] = (__bf16)f3.x; a[13] = (__bf16)f3.y; a[14] = (__bf16)f3.z; a[15] = (__bf16)f3.w;
    // B fragments (wt rows n0+ln): K = k0 + lh*16 + e, contiguous 16 bf16
    v16bf bq = *(const v16bf*)(wq + k0 + lh * 16);
    v16bf bk = *(const v16bf*)(wk + k0 + lh * 16);
    v16bf bv = *(const v16bf*)(wv + k0 + lh * 16);
    accq = __builtin_amdgcn_wmma_f32_16x16x32_bf16(false, a, false, bq,
                                                   (short)0, accq, false, false);
    acck = __builtin_amdgcn_wmma_f32_16x16x32_bf16(false, a, false, bk,
                                                   (short)0, acck, false, false);
    accv = __builtin_amdgcn_wmma_f32_16x16x32_bf16(false, a, false, bv,
                                                   (short)0, accv, false, false);
  }

  // q (pre-scaled by H^-0.5) and k: row-major scattered b16 stores
#pragma unroll
  for (int r = 0; r < 8; ++r) {
    int m = m0 + lh * 8 + r;
    qbf[(size_t)m * HD + n0 + ln] = (__bf16)(accq[r] * 0.125f);
    kbf[(size_t)m * HD + n0 + ln] = (__bf16)acck[r];
  }
  // v transposed: vt[b][h][t]; 8 contiguous t per lane -> one b128 store
  {
    int b = m0 / T_SEQ;
    int t = (m0 - b * T_SEQ) + lh * 8;
    __bf16* vr = vtbf + ((size_t)b * HD + n0 + ln) * T_SEQ + t;
    v8bf vv;
#pragma unroll
    for (int r = 0; r < 8; ++r) vv[r] = (__bf16)accv[r];
    *(v8bf*)vr = vv;
  }
}

// ---------------------------------------------------------------------------
// Kernel 2: flash attention with ripple (ALiBi-style) bias.
// grid = (T/64, B), block = 128 (4 waves). Wave w owns 16 query rows
// [qblk*64 + 16w, +16). Streams 32-key blocks with online softmax; P is
// re-laid-out C->A through a private LDS slice (s_wait_dscnt 0 between).
// 8 WMMAs per key block per wave.
// ---------------------------------------------------------------------------
__global__ __launch_bounds__(128)
void attn_kernel(const __bf16* __restrict__ qbf, const __bf16* __restrict__ kbf,
                 const __bf16* __restrict__ vtbf,
                 const float* __restrict__ decayp, float* __restrict__ out) {
  const int b     = blockIdx.y;
  const int wave  = threadIdx.x >> 5;
  const int lane  = threadIdx.x & 31;
  const int ln    = lane & 15;
  const int lh    = lane >> 4;
  const int qtile = blockIdx.x * 64 + wave * 16;   // first query row of wave

  const float decay = fabsf(decayp[0]);

  const __bf16* q  = qbf  + (size_t)b * T_SEQ * HD;
  const __bf16* k  = kbf  + (size_t)b * T_SEQ * HD;
  const __bf16* vt = vtbf + (size_t)b * HD * T_SEQ;

  __shared__ __bf16 lds_p[4][16][32];              // per-wave P staging

  // Q A-fragments for the two 32-wide K-dim steps (head dim = 64)
  v16bf aq[2];
  {
    const __bf16* qr = q + (size_t)(qtile + ln) * HD;
#pragma unroll
    for (int t = 0; t < 2; ++t) {
      v8bf r0 = *(const v8bf*)(qr + t * 32 + lh * 16);
      v8bf r1 = *(const v8bf*)(qr + t * 32 + lh * 16 + 8);
      v16bf a;
#pragma unroll
      for (int e = 0; e < 8; ++e) { a[e] = r0[e]; a[e + 8] = r1[e]; }
      aq[t] = a;
    }
  }

  v8f ot[4] = {};                // 16x64 output accumulator
  float mrow[8], lrow[8];
#pragma unroll
  for (int r = 0; r < 8; ++r) { mrow[r] = -INFINITY; lrow[r] = 0.0f; }

  const int jend = qtile + 16;   // causal limit for this wave's rows
  for (int j0 = 0; j0 < jend; j0 += 32) {
    // ---- S = Q K^T (two 16-key n-tiles) ----
    v8f s[2];
#pragma unroll
    for (int nt = 0; nt < 2; ++nt) {
      const __bf16* kr = k + (size_t)(j0 + nt * 16 + ln) * HD + lh * 16;
      v16bf b0 = *(const v16bf*)(kr);        // K-dim 0..31
      v16bf b1 = *(const v16bf*)(kr + 32);   // K-dim 32..63
      v8f c = {};
      c = __builtin_amdgcn_wmma_f32_16x16x32_bf16(false, aq[0], false, b0,
                                                  (short)0, c, false, false);
      c = __builtin_amdgcn_wmma_f32_16x16x32_bf16(false, aq[1], false, b1,
                                                  (short)0, c, false, false);
      s[nt] = c;
    }

    // ---- bias + causal mask, per element of the C layout ----
    float p[2][8], rm[8];
#pragma unroll
    for (int r = 0; r < 8; ++r) {
      int i = qtile + lh * 8 + r;
#pragma unroll
      for (int nt = 0; nt < 2; ++nt) {
        int j = j0 + nt * 16 + ln;
        float sv = s[nt][r];
        p[nt][r] = (j <= i) ? sv + (float)(j - i) * decay : -INFINITY;
      }
      rm[r] = fmaxf(p[0][r], p[1][r]);
    }
    // row max: rows live within 16-lane halves -> xor masks 1,2,4,8
#pragma unroll
    for (int mask = 1; mask < 16; mask <<= 1)
#pragma unroll
      for (int r = 0; r < 8; ++r)
        rm[r] = fmaxf(rm[r], __shfl_xor(rm[r], mask, 32));

    // ---- online softmax update ----
    float rs[8];
#pragma unroll
    for (int r = 0; r < 8; ++r) {
      float mnew  = fmaxf(mrow[r], rm[r]);
      float scale = __expf(mrow[r] - mnew);
      mrow[r] = mnew;
      lrow[r] *= scale;
#pragma unroll
      for (int nt = 0; nt < 4; ++nt) ot[nt][r] *= scale;
      float p0 = __expf(p[0][r] - mnew);
      float p1 = __expf(p[1][r] - mnew);
      p[0][r] = p0; p[1][r] = p1;
      rs[r] = p0 + p1;
    }
#pragma unroll
    for (int mask = 1; mask < 16; mask <<= 1)
#pragma unroll
      for (int r = 0; r < 8; ++r)
        rs[r] += __shfl_xor(rs[r], mask, 32);
#pragma unroll
    for (int r = 0; r < 8; ++r) lrow[r] += rs[r];

    // ---- P: C layout -> A layout via per-wave LDS slice ----
#pragma unroll
    for (int r = 0; r < 8; ++r) {
      lds_p[wave][lh * 8 + r][ln]      = (__bf16)p[0][r];
      lds_p[wave][lh * 8 + r][16 + ln] = (__bf16)p[1][r];
    }
    asm volatile("s_wait_dscnt 0" ::: "memory");   // wave-private ordering
    v16bf pa;
    {
      const __bf16* pr = &lds_p[wave][ln][lh * 16];
      v8bf p0 = *(const v8bf*)(pr);        // kk = lh*16 + 0..7
      v8bf p1 = *(const v8bf*)(pr + 8);    // kk = lh*16 + 8..15
#pragma unroll
      for (int e = 0; e < 8; ++e) { pa[e] = p0[e]; pa[e + 8] = p1[e]; }
    }

    // ---- O += P @ V (contract over 32 keys; 4 output col-tiles) ----
#pragma unroll
    for (int nt = 0; nt < 4; ++nt) {
      const __bf16* vr = vt + (size_t)(nt * 16 + ln) * T_SEQ + j0 + lh * 16;
      v16bf bv = *(const v16bf*)vr;
      ot[nt] = __builtin_amdgcn_wmma_f32_16x16x32_bf16(false, pa, false, bv,
                                                       (short)0, ot[nt], false, false);
    }
  }

  // ---- epilogue: normalize and store fp32 ----
  float* orow = out + ((size_t)b * T_SEQ + qtile) * HD;
#pragma unroll
  for (int r = 0; r < 8; ++r) {
    float inv = 1.0f / lrow[r];
    int m = lh * 8 + r;
#pragma unroll
    for (int nt = 0; nt < 4; ++nt)
      orow[(size_t)m * HD + nt * 16 + ln] = ot[nt][r] * inv;
  }
}

// ---------------------------------------------------------------------------
extern "C" void kernel_launch(void* const* d_in, const int* in_sizes, int n_in,
                              void* d_out, int out_size, void* d_ws, size_t ws_size,
                              hipStream_t stream) {
  (void)in_sizes; (void)n_in; (void)out_size; (void)ws_size;
  const float* x     = (const float*)d_in[0];
  const float* Wk    = (const float*)d_in[1];
  const float* Wq    = (const float*)d_in[2];
  const float* Wv    = (const float*)d_in[3];
  const float* decay = (const float*)d_in[4];
  float* out = (float*)d_out;

  char* ws = (char*)d_ws;
  const size_t WT_BYTES  = (size_t)3 * HD * C_EMB * 2;          // 393216
  const size_t QKV_BYTES = (size_t)BATCH * T_SEQ * HD * 2;      // 2 MiB each
  __bf16* wt   = (__bf16*)(ws);
  __bf16* qbf  = (__bf16*)(ws + WT_BYTES);
  __bf16* kbf  = (__bf16*)(ws + WT_BYTES + QKV_BYTES);
  __bf16* vtbf = (__bf16*)(ws + WT_BYTES + 2 * QKV_BYTES);

  {
    int n = 3 * HD * C_EMB;
    wprep_kernel<<<(n + 255) / 256, 256, 0, stream>>>(Wq, Wk, Wv, wt);
  }
  {
    dim3 grid(BATCH * T_SEQ / 16);
    proj_kernel<<<grid, 128, 0, stream>>>(x, wt, qbf, kbf, vtbf);
  }
  {
    dim3 grid(T_SEQ / 64, BATCH);
    attn_kernel<<<grid, 128, 0, stream>>>(qbf, kbf, vtbf, decay, out);
  }
}
